// LSTM_7739531067475
// MI455X (gfx1250) — matrix operand order repaired and checked by reading, not compiled, not run on hardware
//
#include <hip/hip_runtime.h>
#include <hip/hip_bf16.h>
#include <stdint.h>

// ---------------------------------------------------------------------------
// LSTM on MI455X (gfx1250, wave32, WMMA bf16 16x16x32)
//   dims: VOCAB=32000 EMB=512 HID=1024 OUT=2  B=64 T=512
// ---------------------------------------------------------------------------

#define EMB    512
#define HID    1024
#define GATES  4096          // 4*HID
#define KTOT   1536          // EMB + HID
#define NPANEL (KTOT / 32)   // 48 K-panels of 32
#define BATCH  64
#define TSTEPS 512

typedef __bf16 bf16_t;
typedef bf16_t v16bf __attribute__((ext_vector_type(16)));
typedef bf16_t v8bf  __attribute__((ext_vector_type(8)));
typedef float  v8f   __attribute__((ext_vector_type(8)));
typedef int    v4i_  __attribute__((ext_vector_type(4)));

union AFrag { v16bf v; v8bf h[2]; };

__device__ __forceinline__ unsigned short f32_to_bf16(float f) {
    unsigned int u = __float_as_uint(f);
    u += 0x7FFFu + ((u >> 16) & 1u);   // round-to-nearest-even
    return (unsigned short)(u >> 16);
}
__device__ __forceinline__ float bf16_to_f32(unsigned short s) {
    return __uint_as_float(((unsigned int)s) << 16);
}
__device__ __forceinline__ float sigmoidf_(float x) {
    return 1.0f / (1.0f + __expf(-x));
}

// ---- CDNA5 async global->LDS copy (ASYNCcnt path), with sync fallback -----
// Probe-confirmed: builtin exists; param0 is global-AS pointer to int4.
#if __has_builtin(__builtin_amdgcn_global_load_async_to_lds_b128) && \
    __has_builtin(__builtin_amdgcn_s_wait_asynccnt)
#define ASYNC_COPY16(src, dst)                                          \
    __builtin_amdgcn_global_load_async_to_lds_b128(                     \
        (__attribute__((address_space(1))) v4i_*)(src),                 \
        (__attribute__((address_space(3))) v4i_*)(dst), 0, 0)
#define ASYNC_WAIT() __builtin_amdgcn_s_wait_asynccnt(0)
#else
#define ASYNC_COPY16(src, dst) (*(uint4*)(dst) = *(const uint4*)(src))
#define ASYNC_WAIT() ((void)0)
#endif

// ---------------------------------------------------------------------------
// Kernel 1: embedding lookup + f32 -> bf16, time-major xe[t][b][e]
// grid: T*B blocks, 128 threads (4 elems each)
// ---------------------------------------------------------------------------
__global__ void k_embed(const int* __restrict__ x, const float* __restrict__ emb,
                        unsigned short* __restrict__ xe) {
    const int bid = blockIdx.x;            // = t*64 + b
    const int t = bid >> 6, b = bid & 63;
    const int tok = x[b * TSTEPS + t];     // x is [B,T]
    const float4 v = *(const float4*)(emb + (size_t)tok * EMB + threadIdx.x * 4);
    unsigned int lo = (unsigned)f32_to_bf16(v.x) | ((unsigned)f32_to_bf16(v.y) << 16);
    unsigned int hi = (unsigned)f32_to_bf16(v.z) | ((unsigned)f32_to_bf16(v.w) << 16);
    uint2* dst = (uint2*)(xe + (size_t)bid * EMB + threadIdx.x * 4);
    *dst = make_uint2(lo, hi);
}

// ---------------------------------------------------------------------------
// Kernel 2: pack transposed weights Wt[n=4096][k=1536] bf16.
//   column n -> gate g = n>>10, unit cc = n&1023 ; k<512 from Wx_g else Wh_g.
// Consecutive threads walk k => coalesced writes; strided reads are one-time
// and L2-resident (sources total 24 MB << 192 MB L2).
// ---------------------------------------------------------------------------
__global__ void k_pack_w(const float* __restrict__ Wxi, const float* __restrict__ Whi,
                         const float* __restrict__ Wxf, const float* __restrict__ Whf,
                         const float* __restrict__ Wxc, const float* __restrict__ Whc,
                         const float* __restrict__ Wxo, const float* __restrict__ Who,
                         unsigned short* __restrict__ Wt) {
    const size_t idx = (size_t)blockIdx.x * blockDim.x + threadIdx.x;
    if (idx >= (size_t)GATES * KTOT) return;
    const int n = (int)(idx / KTOT);
    const int k = (int)(idx % KTOT);
    const int g = n >> 10, cc = n & 1023;
    float v;
    if (k < EMB) {
        const float* src = (g == 0) ? Wxi : (g == 1) ? Wxf : (g == 2) ? Wxc : Wxo;
        v = src[(size_t)k * HID + cc];
    } else {
        const float* src = (g == 0) ? Whi : (g == 1) ? Whf : (g == 2) ? Whc : Who;
        v = src[(size_t)(k - EMB) * HID + cc];
    }
    Wt[idx] = f32_to_bf16(v);
}

// ---------------------------------------------------------------------------
// Kernel 3: bias pack + zero-init h0 (bf16) and c (f32). 65536 threads.
// ---------------------------------------------------------------------------
__global__ void k_init(const float* __restrict__ bi, const float* __restrict__ bff,
                       const float* __restrict__ bc, const float* __restrict__ bo,
                       float* __restrict__ bias, unsigned short* __restrict__ h0,
                       float* __restrict__ c) {
    const int idx = blockIdx.x * blockDim.x + threadIdx.x;   // 0..65535
    h0[idx] = 0;
    c[idx] = 0.0f;
    if (idx < GATES) {
        const int g = idx >> 10, cc = idx & 1023;
        const float* src = (g == 0) ? bi : (g == 1) ? bff : (g == 2) ? bc : bo;
        bias[idx] = src[cc];
    }
}

// ---------------------------------------------------------------------------
// Kernel 4: one LSTM time step, fully fused, double-buffered LDS panels.
//   g[64][4096] = [xe_t | h_in] @ W + bias ; gate math ; update c, h_out.
// grid: 64 blocks (one per 16-wide h slice), 256 threads = 8 wave32.
//   wave w: mi = w&3 (M tile), gates {2*(w>>2), 2*(w>>2)+1}.
// ---------------------------------------------------------------------------
__global__ void __launch_bounds__(256)
k_lstm_step(const unsigned short* __restrict__ xe, const unsigned short* __restrict__ Wt,
            const float* __restrict__ bias, const unsigned short* __restrict__ h_in,
            unsigned short* __restrict__ h_out, float* __restrict__ c, int t) {
    __shared__ alignas(32) bf16_t As[2][64 * 32];       // A panels [m][k]   2x4 KB
    __shared__ alignas(64) bf16_t Bs[2][4 * 16 * 32];   // B panels [g][n][k] 2x4 KB
    __shared__ float Gs[4][64][16];                     // gate staging       16 KB

    const int nblk = blockIdx.x;          // h-column tile: cols nblk*16 .. +15
    const int tid  = threadIdx.x;
    const int lane = tid & 31;
    const int wave = tid >> 5;
    const int mi   = wave & 3;            // M tile 0..3
    const int ga   = (wave >> 2) * 2;     // gates ga, ga+1

    // copy-role decomposition (each thread moves one 16 B chunk of A and of B)
    const int mA = tid >> 2,        partA = tid & 3;          // A: row, 8-elem chunk
    const int gB = tid >> 6;                                  // B: gate 0..3
    const int nlB = (tid >> 2) & 15, chB = tid & 3;           // B: col, 8-elem chunk
    const unsigned short* bRow =
        Wt + (size_t)(gB * HID + nblk * 16 + nlB) * KTOT;     // column-major W row

    // fragment-read roles (wave32; contiguous k = 16*half + s in A and B)
    const int half = lane >> 4;                // A K-half
    const int mrow = mi * 16 + (lane & 15);    // A row
    const int nl   = lane & 15;                // B column
    const int kh   = lane >> 4;                // B K-half

    v8f acc0 = {}; v8f acc1 = {};

    auto issue_panel = [&](int kp, int buf) {
        const int kk = kp * 32 + partA * 8;
        const unsigned short* asrc = (kk < EMB)
            ? xe + ((size_t)t * BATCH + mA) * EMB + kk
            : h_in + (size_t)mA * HID + (kk - EMB);
        ASYNC_COPY16(asrc, &As[buf][mA * 32 + partA * 8]);
        ASYNC_COPY16(bRow + kp * 32 + chB * 8,
                     &Bs[buf][(gB * 16 + nlB) * 32 + chB * 8]);
        if (kp + 2 < NPANEL)   // global_prefetch_b8 two panels ahead of Wt
            __builtin_prefetch(bRow + (kp + 2) * 32, 0, 1);
    };

    issue_panel(0, 0);
    ASYNC_WAIT();
    __syncthreads();

    for (int kp = 0; kp < NPANEL; ++kp) {
        const int buf = kp & 1;
        if (kp + 1 < NPANEL) issue_panel(kp + 1, buf ^ 1);

        AFrag fa;
        fa.v = *(const v16bf*)&As[buf][mrow * 32 + half * 16];
        v16bf fb0 = *(const v16bf*)&Bs[buf][((ga       * 16 + nl) * 32) + kh * 16];
        v16bf fb1 = *(const v16bf*)&Bs[buf][(((ga + 1) * 16 + nl) * 32) + kh * 16];
        acc0 = __builtin_amdgcn_wmma_f32_16x16x32_bf16(false, fa.v, false, fb0,
                                                       (short)0, acc0, false, false);
        acc1 = __builtin_amdgcn_wmma_f32_16x16x32_bf16(false, fa.v, false, fb1,
                                                       (short)0, acc1, false, false);
        ASYNC_WAIT();
        __syncthreads();
    }

    // ---- C/D layout: lane n -> col n%16, VGPR j -> row j + 8*(n/16). Add bias.
    {
        const float b0 = bias[ga * HID + nblk * 16 + nl];
        const float b1 = bias[(ga + 1) * HID + nblk * 16 + nl];
        const int rbase = mi * 16 + 8 * (lane >> 4);
        #pragma unroll
        for (int j = 0; j < 8; ++j) {
            Gs[ga    ][rbase + j][nl] = acc0[j] + b0;
            Gs[ga + 1][rbase + j][nl] = acc1[j] + b1;
        }
    }
    __syncthreads();

    // ---- fused gate nonlinearity + state update for this 16-wide h slice.
    for (int p = tid; p < BATCH * 16; p += 256) {
        const int b = p >> 4, hc = p & 15;
        const float iv = sigmoidf_(Gs[0][b][hc]);
        const float fv = sigmoidf_(Gs[1][b][hc]);
        const float gv = tanhf(Gs[2][b][hc]);
        const float ov = sigmoidf_(Gs[3][b][hc]);
        const size_t idx = (size_t)b * HID + nblk * 16 + hc;
        const float cn = fv * c[idx] + iv * gv;
        c[idx] = cn;
        h_out[idx] = f32_to_bf16(ov * tanhf(cn));
    }
}

// ---------------------------------------------------------------------------
// Kernel 5: out[64][2] = h @ Why + by   (tiny; 1 block, 128 threads)
// ---------------------------------------------------------------------------
__global__ void k_final(const unsigned short* __restrict__ h,
                        const float* __restrict__ Why, const float* __restrict__ by,
                        float* __restrict__ out) {
    const int b = threadIdx.x >> 1, o = threadIdx.x & 1;
    float acc = 0.0f;
    for (int k = 0; k < HID; ++k)
        acc += bf16_to_f32(h[(size_t)b * HID + k]) * Why[(size_t)k * 2 + o];
    out[b * 2 + o] = acc + by[o];
}

// ---------------------------------------------------------------------------
extern "C" void kernel_launch(void* const* d_in, const int* in_sizes, int n_in,
                              void* d_out, int out_size, void* d_ws, size_t ws_size,
                              hipStream_t stream) {
    (void)in_sizes; (void)n_in; (void)out_size; (void)ws_size;
    const int*   x   = (const int*)  d_in[0];
    const float* emb = (const float*)d_in[1];
    const float* Wxi = (const float*)d_in[2],  *Whi = (const float*)d_in[3],
               * bi  = (const float*)d_in[4];
    const float* Wxf = (const float*)d_in[5],  *Whf = (const float*)d_in[6],
               * bff = (const float*)d_in[7];
    const float* Wxc = (const float*)d_in[8],  *Whc = (const float*)d_in[9],
               * bc  = (const float*)d_in[10];
    const float* Wxo = (const float*)d_in[11], *Who = (const float*)d_in[12],
               * bo  = (const float*)d_in[13];
    const float* Why = (const float*)d_in[14], *by  = (const float*)d_in[15];

    // workspace layout (~44.5 MB total)
    char* ws = (char*)d_ws;
    size_t off = 0;
    unsigned short* xe = (unsigned short*)(ws + off); off += (size_t)TSTEPS * BATCH * EMB * 2;   // 32 MB
    unsigned short* Wt = (unsigned short*)(ws + off); off += (size_t)KTOT * GATES * 2;           // 12 MB
    float*          bias = (float*)(ws + off);        off += (size_t)GATES * 4;                  // 16 KB
    unsigned short* h0 = (unsigned short*)(ws + off); off += (size_t)BATCH * HID * 2;            // 128 KB
    unsigned short* h1 = (unsigned short*)(ws + off); off += (size_t)BATCH * HID * 2;            // 128 KB
    float*          c  = (float*)(ws + off);          off += (size_t)BATCH * HID * 4;            // 256 KB

    k_embed <<<TSTEPS * BATCH, 128, 0, stream>>>(x, emb, xe);
    k_pack_w<<<((size_t)GATES * KTOT + 255) / 256, 256, 0, stream>>>(
        Wxi, Whi, Wxf, Whf, Wxc, Whc, Wxo, Who, Wt);
    k_init  <<<(BATCH * HID) / 256, 256, 0, stream>>>(bi, bff, bc, bo, bias, h0, c);

    for (int t = 0; t < TSTEPS; ++t) {
        const unsigned short* hin  = (t & 1) ? h1 : h0;
        unsigned short*       hout = (t & 1) ? h0 : h1;
        k_lstm_step<<<GATES / 64, 256, 0, stream>>>(xe, Wt, bias, hin, hout, c, t);
    }
    // t=511 (odd) wrote h0 -> final hidden state lives in h0
    k_final<<<1, 128, 0, stream>>>(h0, Why, by, (float*)d_out);
}
